// Attention_20246475833764
// MI455X (gfx1250) — matrix operand order repaired and checked by reading, not compiled, not run on hardware
//
#include <hip/hip_runtime.h>
#include <math.h>

// ---------------------------------------------------------------------------
// Attention forward for MI455X (gfx1250, wave32, WMMA).
//   B=1, S=2048, D=4096, H=32, KVH=8, HD=128, N_REP=4, MULT=1
// Pipeline: f32->bf16 cvt, QKV GEMMs (WMMA bf16/f32-acc), RoPE(+2^-7 on Q),
// V transpose, flash attention (WMMA), O projection GEMM -> d_out (f32).
// LDS staging uses gfx1250 async global->LDS (ASYNCcnt) with double buffering.
// Workspace use: ~168 MiB in d_ws.
// ---------------------------------------------------------------------------

#define S_LEN    2048
#define D_MODEL  4096
#define N_HEADS  32
#define N_KVH    8
#define HEAD_DIM 128

typedef __bf16 bf16_t;
typedef __attribute__((ext_vector_type(16))) __bf16 v16bf;
typedef __attribute__((ext_vector_type(8)))  __bf16 v8bf;
typedef __attribute__((ext_vector_type(4)))  __bf16 v4bf;
typedef __attribute__((ext_vector_type(8)))  float  v8f;
// exact pointee type of the async-to-LDS builtins (per clang diagnostic)
typedef int b128i __attribute__((vector_size(16)));

// ------------------- gfx1250 async-to-LDS path (guarded) --------------------
#if defined(__AMDGCN__)
#if __has_builtin(__builtin_amdgcn_global_load_async_to_lds_b128) && \
    __has_builtin(__builtin_amdgcn_s_wait_asynccnt)
#define HAVE_ASYNC_LDS 1
#endif
#endif

// Copy 16B global -> LDS. Async (no VGPR round trip, ASYNCcnt) when available.
__device__ __forceinline__ void stage_b128(const bf16_t* g, bf16_t* l) {
#if defined(HAVE_ASYNC_LDS)
  // generic->AS1 is value-preserving; generic->AS3 truncates to LDS offset.
  __builtin_amdgcn_global_load_async_to_lds_b128(
      (__attribute__((address_space(1))) b128i*)(unsigned long long)g,
      (__attribute__((address_space(3))) b128i*)(unsigned int)(unsigned long long)l,
      0, 0);
#else
  *(v8bf*)l = *(const v8bf*)g;
#endif
}

__device__ __forceinline__ void stage_wait_all() {
#if defined(HAVE_ASYNC_LDS)
  __builtin_amdgcn_s_wait_asynccnt(0);
#endif
}

// ------------------- small device helpers ----------------------------------
__device__ __forceinline__ v16bf bf_cat(v8bf lo, v8bf hi) {
  return __builtin_shufflevector(lo, hi, 0, 1, 2, 3, 4, 5, 6, 7,
                                 8, 9, 10, 11, 12, 13, 14, 15);
}

__device__ __forceinline__ v8f wmma_bf16(v16bf a, v16bf b, v8f c) {
  // D(f32 16x16) = A(16x32 bf16) * B(32x16 bf16) + C
  return __builtin_amdgcn_wmma_f32_16x16x32_bf16(false, a, false, b,
                                                 (short)0, c, false, false);
}

// ------------------- fp32 -> bf16 bulk convert ------------------------------
__global__ __launch_bounds__(256) void cvt_f32_to_bf16(
    const float* __restrict__ in, bf16_t* __restrict__ out, long n4) {
  const long i = (long)blockIdx.x * blockDim.x + threadIdx.x;
  if (i < n4) {
    const float4 v = *(const float4*)(in + i * 4);
    v4bf o;
    o[0] = (bf16_t)v.x;
    o[1] = (bf16_t)v.y;
    o[2] = (bf16_t)v.z;
    o[3] = (bf16_t)v.w;
    *(v4bf*)(out + i * 4) = o;
  }
}

// ------------------- GEMM: C[M,N](f32) = A[M,K](bf16) * W[N,K]^T(bf16) -----
// 128x128 block tile, 8 waves (4 along M x 2 along N), wave tile 32x64,
// K-step 32, double-buffered async LDS staging.
#define GSTR 40  // LDS row stride in bf16 (32 data + 8 pad) = 80 B, 16B aligned

__global__ __launch_bounds__(256) void gemm_xwT_bf16_f32(
    const bf16_t* __restrict__ A, const bf16_t* __restrict__ W,
    float* __restrict__ C, int M, int N, int K) {
  __shared__ __align__(16) bf16_t Al[2][128 * GSTR];
  __shared__ __align__(16) bf16_t Wl[2][128 * GSTR];

  const int tid = threadIdx.x;
  const int lane = tid & 31;
  const int wave = tid >> 5;
  const int wm = wave & 3;    // 4 waves along M
  const int wn = wave >> 2;   // 2 waves along N
  const int lcol = lane & 15;
  const int lhalf = lane >> 4;

  const int blockM = blockIdx.x * 128;
  const int blockN = blockIdx.y * 128;

  v8f acc[2][4];
#pragma unroll
  for (int i = 0; i < 2; ++i)
#pragma unroll
    for (int j = 0; j < 4; ++j)
      acc[i][j] = (v8f){0.f, 0.f, 0.f, 0.f, 0.f, 0.f, 0.f, 0.f};

  const int sr = tid >> 1;         // staging row 0..127
  const int sc0 = (tid & 1) * 16;  // staging col 0 or 16
  const bf16_t* gArow = A + (size_t)(blockM + sr) * K + sc0;
  const bf16_t* gWrow = W + (size_t)(blockN + sr) * K + sc0;

  auto stage = [&](int buf, int k0) {
    stage_b128(gArow + k0, &Al[buf][sr * GSTR + sc0]);
    stage_b128(gArow + k0 + 8, &Al[buf][sr * GSTR + sc0 + 8]);
    stage_b128(gWrow + k0, &Wl[buf][sr * GSTR + sc0]);
    stage_b128(gWrow + k0 + 8, &Wl[buf][sr * GSTR + sc0 + 8]);
    __builtin_prefetch(gArow + k0 + 32, 0, 1);  // warm L2 one slab ahead
    __builtin_prefetch(gWrow + k0 + 32, 0, 1);
  };

  const int nsteps = K >> 5;
  stage(0, 0);
  for (int i = 0; i < nsteps; ++i) {
    const int buf = i & 1;
    stage_wait_all();   // this wave's fills of Al/Wl[buf] complete
    __syncthreads();    // everyone's fills complete; prev compute done
    if (i + 1 < nsteps) stage(buf ^ 1, (i + 1) << 5);

    const bf16_t* Ab = &Al[buf][0];
    const bf16_t* Wb = &Wl[buf][0];
    // A fragments (16x32): lane holds row (lane&15); K halves per lane-half.
    v16bf af[2];
#pragma unroll
    for (int r = 0; r < 2; ++r) {
      const bf16_t* pa = Ab + (wm * 32 + r * 16 + lcol) * GSTR;
      af[r] = bf_cat(*(const v8bf*)(pa + lhalf * 8),
                     *(const v8bf*)(pa + 16 + lhalf * 8));
    }
    // B fragments (32x16): lane holds column (lane&15); K halves per lane-half.
#pragma unroll
    for (int j = 0; j < 4; ++j) {
      const bf16_t* pb = Wb + (wn * 64 + j * 16 + lcol) * GSTR;
      v16bf bfr = bf_cat(*(const v8bf*)(pb + lhalf * 8), *(const v8bf*)(pb + 16 + lhalf * 8));
      acc[0][j] = wmma_bf16(af[0], bfr, acc[0][j]);
      acc[1][j] = wmma_bf16(af[1], bfr, acc[1][j]);
    }
  }

  // C layout: VGPR v, lanes 0-15 -> row v, lanes 16-31 -> row v+8; col=lane&15
#pragma unroll
  for (int r = 0; r < 2; ++r) {
    const int row0 = blockM + wm * 32 + r * 16 + lhalf * 8;
#pragma unroll
    for (int j = 0; j < 4; ++j) {
      const int col = blockN + wn * 64 + j * 16 + lcol;
#pragma unroll
      for (int v = 0; v < 8; ++v)
        C[(size_t)(row0 + v) * N + col] = acc[r][j][v];
    }
  }
}

// ------------------- RoPE (+optional scale) f32 -> bf16 ---------------------
__global__ __launch_bounds__(64) void rope_rows_bf16(
    const float* __restrict__ in, bf16_t* __restrict__ out, int nheads,
    float scale) {
  const int s = blockIdx.x, h = blockIdx.y, i = threadIdx.x;  // i in [0,64)
  const size_t base = ((size_t)s * nheads + h) * HEAD_DIM;
  const float freq = __powf(10000.f, -(float)i * (1.f / 64.f));
  const float ang = (float)s * freq;
  float sn, cs;
  __sincosf(ang, &sn, &cs);
  const float x0 = in[base + i];
  const float x1 = in[base + i + 64];
  out[base + i] = (bf16_t)((cs * x0 - sn * x1) * scale);
  out[base + i + 64] = (bf16_t)((cs * x1 + sn * x0) * scale);
}

// ------------------- V transpose: V[S,1024](f32) -> Vt[1024,S](bf16) --------
__global__ __launch_bounds__(256) void transpose_v_bf16(
    const float* __restrict__ V, bf16_t* __restrict__ Vt) {
  const int s = blockIdx.x * blockDim.x + threadIdx.x;  // coalesced writes
  const int dc = blockIdx.y;                            // 0..1023
  Vt[(size_t)dc * S_LEN + s] = (bf16_t)V[(size_t)s * (N_KVH * HEAD_DIM) + dc];
}

// ------------------- flash attention (causal, GQA) --------------------------
// grid = (S/128 query tiles, H). 8 waves/block, each wave owns 16 query rows.
// Per 32-key step: async-stage K (32x128) and Vt (128x32) into the off buffer,
// S=Q*K^T via 2x4 WMMAs, online softmax (shfl width-16 row reductions),
// P->LDS->A-frag, O += P*V via 8 WMMAs.
#define KBLK 32
#define KL_STR (HEAD_DIM + 8)  // 136 bf16 = 272 B
#define VL_STR (KBLK + 8)      // 40 bf16  = 80 B

__global__ __launch_bounds__(256) void flash_attn_bf16(
    const bf16_t* __restrict__ Q,   // [S][H*HD], pre-scaled by 1/HD
    const bf16_t* __restrict__ Kc,  // [S][KVH*HD]
    const bf16_t* __restrict__ Vt,  // [KVH*HD][S]
    bf16_t* __restrict__ O,         // [S][H*HD]
    float out_scale) {
  __shared__ __align__(16) bf16_t Kl[2][KBLK * KL_STR];      // 2 x 8704 B
  __shared__ __align__(16) bf16_t Vl[2][HEAD_DIM * VL_STR];  // 2 x 10240 B
  __shared__ __align__(16) bf16_t Pl[8 * 16 * VL_STR];       // 10240 B

  const int tid = threadIdx.x, lane = tid & 31, wave = tid >> 5;
  const int lcol = lane & 15, lhalf = lane >> 4;
  const int h = blockIdx.y, hkv = h >> 2;  // N_REP = 4
  const int q0 = blockIdx.x * 128;
  const int qr = q0 + wave * 16;

  // Q fragments kept in registers: 4 chunks of 32 over head dim
  v16bf qf[4];
  const bf16_t* qrow = Q + (size_t)(qr + lcol) * D_MODEL + h * HEAD_DIM;
#pragma unroll
  for (int c = 0; c < 4; ++c)
    qf[c] = bf_cat(*(const v8bf*)(qrow + c * 32 + lhalf * 8),
                   *(const v8bf*)(qrow + c * 32 + 16 + lhalf * 8));

  v8f o[8];
#pragma unroll
  for (int d = 0; d < 8; ++d)
    o[d] = (v8f){0.f, 0.f, 0.f, 0.f, 0.f, 0.f, 0.f, 0.f};
  float mrow[8], lrow[8];
#pragma unroll
  for (int v = 0; v < 8; ++v) {
    mrow[v] = -1e30f;
    lrow[v] = 0.f;
  }

  // staging indices
  const int kr = tid >> 3, kc = (tid & 7) * 16;  // K: 32 rows x 128 cols
  const int vr = tid >> 1, vc = (tid & 1) * 16;  // V: 128 rows x 32 cols
  const bf16_t* gKrow = Kc + (size_t)kr * (N_KVH * HEAD_DIM) + hkv * HEAD_DIM + kc;
  const bf16_t* gVrow = Vt + ((size_t)hkv * HEAD_DIM + vr) * S_LEN + vc;

  auto stage = [&](int buf, int kbase) {
    const bf16_t* gk = gKrow + (size_t)kbase * (N_KVH * HEAD_DIM);
    stage_b128(gk, &Kl[buf][kr * KL_STR + kc]);
    stage_b128(gk + 8, &Kl[buf][kr * KL_STR + kc + 8]);
    const bf16_t* gv = gVrow + kbase;
    stage_b128(gv, &Vl[buf][vr * VL_STR + vc]);
    stage_b128(gv + 8, &Vl[buf][vr * VL_STR + vc + 8]);
    __builtin_prefetch(gk + (size_t)KBLK * (N_KVH * HEAD_DIM), 0, 1);
    __builtin_prefetch(gv + KBLK, 0, 1);
  };

  const int ksteps = (q0 + 128) >> 5;  // causal: keys < q0+128
  stage(0, 0);
  for (int ks = 0; ks < ksteps; ++ks) {
    const int buf = ks & 1;
    const int kbase = ks * KBLK;
    stage_wait_all();
    __syncthreads();
    if (ks + 1 < ksteps) stage(buf ^ 1, kbase + KBLK);

    const bf16_t* Kb = &Kl[buf][0];
    const bf16_t* Vb = &Vl[buf][0];

    // scores: two 16-key tiles, 4 head-dim chunks each
    v8f sc[2];
    sc[0] = (v8f){0.f, 0.f, 0.f, 0.f, 0.f, 0.f, 0.f, 0.f};
    sc[1] = sc[0];
#pragma unroll
    for (int t = 0; t < 2; ++t)
#pragma unroll
      for (int c = 0; c < 4; ++c) {
        const bf16_t* pk = Kb + (t * 16 + lcol) * KL_STR + c * 32;
        v16bf kf = bf_cat(*(const v8bf*)(pk + lhalf * 8), *(const v8bf*)(pk + 16 + lhalf * 8));
        sc[t] = wmma_bf16(qf[c], kf, sc[t]);
      }

    // causal mask + online softmax update (row = v + 8*lhalf, col = lane&15)
#pragma unroll
    for (int v = 0; v < 8; ++v) {
      const int rg = qr + v + lhalf * 8;
      float s0 = ((kbase + lcol) > rg) ? -1e30f : sc[0][v];
      float s1 = ((kbase + 16 + lcol) > rg) ? -1e30f : sc[1][v];
      float mt = fmaxf(s0, s1);
      mt = fmaxf(mt, __shfl_xor(mt, 1, 16));
      mt = fmaxf(mt, __shfl_xor(mt, 2, 16));
      mt = fmaxf(mt, __shfl_xor(mt, 4, 16));
      mt = fmaxf(mt, __shfl_xor(mt, 8, 16));
      const float mn = fmaxf(mrow[v], mt);
      const float al = __expf(mrow[v] - mn);
      const float p0 = __expf(s0 - mn);
      const float p1 = __expf(s1 - mn);
      float rs = p0 + p1;
      rs += __shfl_xor(rs, 1, 16);
      rs += __shfl_xor(rs, 2, 16);
      rs += __shfl_xor(rs, 4, 16);
      rs += __shfl_xor(rs, 8, 16);
      lrow[v] = lrow[v] * al + rs;
      mrow[v] = mn;
      sc[0][v] = p0;
      sc[1][v] = p1;
#pragma unroll
      for (int d = 0; d < 8; ++d) o[d][v] *= al;
    }

    // C-layout P -> per-wave LDS tile -> A-layout fragment (wave-private)
    bf16_t* pw = &Pl[wave * 16 * VL_STR];
#pragma unroll
    for (int v = 0; v < 8; ++v) {
      const int rr = v + lhalf * 8;
      pw[rr * VL_STR + lcol] = (bf16_t)sc[0][v];
      pw[rr * VL_STR + 16 + lcol] = (bf16_t)sc[1][v];
    }
    const bf16_t* pp = &pw[lcol * VL_STR];
    v16bf pf = bf_cat(*(const v8bf*)(pp + lhalf * 8),
                      *(const v8bf*)(pp + 16 + lhalf * 8));
#pragma unroll
    for (int d = 0; d < 8; ++d) {
      const bf16_t* pv = Vb + (d * 16 + lcol) * VL_STR;
      v16bf vf = bf_cat(*(const v8bf*)(pv + lhalf * 8), *(const v8bf*)(pv + 16 + lhalf * 8));
      o[d] = wmma_bf16(pf, vf, o[d]);
    }
    // next top-of-loop barrier protects buffer reuse
  }

  // epilogue: O * out_scale / rowsum -> bf16, layout [S][H*HD]
#pragma unroll
  for (int v = 0; v < 8; ++v) {
    const float inv = out_scale / lrow[v];
    bf16_t* orow = O + (size_t)(qr + v + lhalf * 8) * D_MODEL + h * HEAD_DIM;
#pragma unroll
    for (int d = 0; d < 8; ++d)
      orow[d * 16 + lcol] = (bf16_t)(o[d][v] * inv);
  }
}

// ------------------- host orchestration -------------------------------------
extern "C" void kernel_launch(void* const* d_in, const int* in_sizes, int n_in,
                              void* d_out, int out_size, void* d_ws,
                              size_t ws_size, hipStream_t stream) {
  const float* x = (const float*)d_in[0];
  // d_in[1] = freqs_cis: unused (reference recomputes RoPE tables from S, HD)
  const float* wq = (const float*)d_in[2];
  const float* wk = (const float*)d_in[3];
  const float* wv = (const float*)d_in[4];
  const float* wo = (const float*)d_in[5];
  float* out = (float*)d_out;

  char* p = (char*)d_ws;
  auto take = [&](size_t bytes) -> char* {
    char* r = p;
    p += (bytes + 255) & ~(size_t)255;
    return r;
  };
  const size_t SD = (size_t)S_LEN * D_MODEL;            // 8.39M elems
  const size_t SKV = (size_t)S_LEN * N_KVH * HEAD_DIM;  // 2.10M elems
  bf16_t* xb = (bf16_t*)take(SD * 2);
  bf16_t* wqb = (bf16_t*)take((size_t)D_MODEL * D_MODEL * 2);
  bf16_t* wkb = (bf16_t*)take((size_t)N_KVH * HEAD_DIM * D_MODEL * 2);
  bf16_t* wvb = (bf16_t*)take((size_t)N_KVH * HEAD_DIM * D_MODEL * 2);
  bf16_t* wob = (bf16_t*)take((size_t)D_MODEL * D_MODEL * 2);
  float* Qf = (float*)take(SD * 4);
  float* Kf = (float*)take(SKV * 4);
  float* Vf = (float*)take(SKV * 4);
  bf16_t* Qbb = (bf16_t*)take(SD * 2);
  bf16_t* Kbb = (bf16_t*)take(SKV * 2);
  bf16_t* Vtb = (bf16_t*)take(SKV * 2);
  bf16_t* AOb = (bf16_t*)Qf;  // reuse: Qf dead after RoPE

  // attn output scale (identical math to reference _attn_scale, MULT=1)
  const double alpha = 1.0 / (1.0 + 4.0 * HEAD_DIM);
  const double lower = sqrt(log((double)S_LEN) / (double)S_LEN);
  const float attn_scale = (float)exp(-(1.0 - alpha) * log(lower));

  const int thr = 256;
  auto cvt = [&](const float* src, bf16_t* dst, size_t n) {
    const long n4 = (long)(n / 4);
    cvt_f32_to_bf16<<<dim3((unsigned)((n4 + thr - 1) / thr)), dim3(thr), 0,
                      stream>>>(src, dst, n4);
  };
  cvt(x, xb, SD);
  cvt(wq, wqb, (size_t)D_MODEL * D_MODEL);
  cvt(wk, wkb, (size_t)N_KVH * HEAD_DIM * D_MODEL);
  cvt(wv, wvb, (size_t)N_KVH * HEAD_DIM * D_MODEL);
  cvt(wo, wob, (size_t)D_MODEL * D_MODEL);

  // QKV projections: out[M,N] = x[M,K] * w[N,K]^T
  gemm_xwT_bf16_f32<<<dim3(S_LEN / 128, D_MODEL / 128), dim3(256), 0, stream>>>(
      xb, wqb, Qf, S_LEN, D_MODEL, D_MODEL);
  gemm_xwT_bf16_f32<<<dim3(S_LEN / 128, (N_KVH * HEAD_DIM) / 128), dim3(256), 0,
                      stream>>>(xb, wkb, Kf, S_LEN, N_KVH * HEAD_DIM, D_MODEL);
  gemm_xwT_bf16_f32<<<dim3(S_LEN / 128, (N_KVH * HEAD_DIM) / 128), dim3(256), 0,
                      stream>>>(xb, wvb, Vf, S_LEN, N_KVH * HEAD_DIM, D_MODEL);

  // RoPE (+fold 1/HD = 2^-7 into Q, exact in bf16) and V transpose
  rope_rows_bf16<<<dim3(S_LEN, N_HEADS), dim3(64), 0, stream>>>(
      Qf, Qbb, N_HEADS, 1.0f / (float)HEAD_DIM);
  rope_rows_bf16<<<dim3(S_LEN, N_KVH), dim3(64), 0, stream>>>(Kf, Kbb, N_KVH,
                                                              1.0f);
  transpose_v_bf16<<<dim3(S_LEN / 256, N_KVH * HEAD_DIM), dim3(256), 0,
                     stream>>>(Vf, Vtb);

  // flash attention -> AOb [S][H*HD] bf16 (attn_scale folded into epilogue)
  flash_attn_bf16<<<dim3(S_LEN / 128, N_HEADS), dim3(256), 0, stream>>>(
      Qbb, Kbb, Vtb, AOb, attn_scale);

  // output projection -> d_out (f32)
  gemm_xwT_bf16_f32<<<dim3(S_LEN / 128, D_MODEL / 128), dim3(256), 0, stream>>>(
      AOb, wob, out, S_LEN, D_MODEL, D_MODEL);
}
